// MultilayerGRU_59425167507907
// MI455X (gfx1250) — matrix operand order repaired and checked by reading, not compiled, not run on hardware
//
#include <hip/hip_runtime.h>

// ---------------------------------------------------------------------------
// Multilayer GRU (B=64, S=1024, I=H=512, L=3, O=128) for gfx1250 (CDNA5).
//
// grid = 4 workgroups (one 16-row batch tile each), block = 512 (16 wave32).
// Per step/layer, gates computed with V_WMMA_F32_16X16X32_BF16:
//   - z & r share A fragments (xact, hbuf): 2 independent accumulator chains
//     per kt iteration -> half the ds traffic, WMMA RAW hazards hidden.
//   - g pass: both of a wave's nt tiles share A (xact, rhbuf).
// Hidden-state master kept in bf16 in global scratch and staged into LDS via
// GLOBAL_LOAD_ASYNC_TO_LDS_B128 (ASYNCcnt) each layer-step.
// Weight tiles are bf16 B-fragment packed (L2-resident, ~9 MB) and prefetched
// with global_prefetch_b8.
// ---------------------------------------------------------------------------

#define B_ 64
#define S_ 1024
#define I_ 512
#define H_ 512
#define L_ 3
#define O_ 128

typedef __attribute__((ext_vector_type(16))) __bf16 v16bf;
typedef __attribute__((ext_vector_type(8)))  __bf16 v8bf;
typedef __attribute__((ext_vector_type(8)))  float  v8f;

// Packed-weight workspace (bf16): [type 0..5][layer 0..2][H*H] then Why[H*O].
// Followed by bf16 hstate[L][B][H].
#define WELEMS_PER (H_ * H_)            // 262144
#define NMAT       (6 * L_)             // 18
#define WHY_OFF    ((size_t)NMAT * WELEMS_PER)
#define WHY_ELEMS  (H_ * O_)
#define PACK_TOTAL (WHY_OFF + WHY_ELEMS)
#define HSTATE_OFF_BYTES ((size_t)PACK_TOTAL * 2)   // 32B aligned

// ---------------------------------------------------------------------------
// B-fragment packing: WMMA 16x16x32 bf16 B (32x16): lane l -> column n=l&15,
// K-half l>>4, element e <-> K=(l>>4)*8+(e&7)+(e>>3)*16. Lane data stored contiguously:
//   dst[((kt*NT+nt)*32 + l)*16 + e]
// ---------------------------------------------------------------------------
__global__ void pack_weights_kernel(const float* __restrict__ Wxz, const float* __restrict__ Whz,
                                    const float* __restrict__ Wxr, const float* __restrict__ Whr,
                                    const float* __restrict__ Wxg, const float* __restrict__ Whg,
                                    const float* __restrict__ Why, __bf16* __restrict__ dst) {
  size_t idx = (size_t)blockIdx.x * blockDim.x + threadIdx.x;
  if (idx >= PACK_TOTAL) return;
  const float* srcs[6] = {Wxz, Whz, Wxr, Whr, Wxg, Whg};
  float v;
  if (idx < WHY_OFF) {
    int mat  = (int)(idx / WELEMS_PER);     // 0..17
    int off  = (int)(idx % WELEMS_PER);
    int type = mat / L_, layer = mat % L_;
    int T  = off >> 9;                      // tile = kt*32 + nt
    int l  = (off >> 4) & 31;
    int e  = off & 15;
    int kt = T >> 5, nt = T & 31;           // N/16 = 32
    int n  = nt * 16 + (l & 15);
    int k  = kt * 32 + ((l >> 4) << 3) + (e & 7) + ((e >> 3) << 4);
    v = srcs[type][(size_t)layer * WELEMS_PER + (size_t)k * H_ + n];
  } else {
    int off = (int)(idx - WHY_OFF);
    int T  = off >> 9;
    int l  = (off >> 4) & 31;
    int e  = off & 15;
    int kt = T >> 3, nt = T & 7;            // N/16 = 8 (O=128)
    int n  = nt * 16 + (l & 15);
    int k  = kt * 32 + ((l >> 4) << 3) + (e & 7) + ((e >> 3) << 4);
    v = Why[(size_t)k * O_ + n];
  }
  dst[idx] = (__bf16)v;
}

// ---------------------------------------------------------------------------
// Fragment helpers
// ---------------------------------------------------------------------------

// A fragment (16x32 bf16) from row-major LDS [16][512]:
// lanes 0-15 (M=l): K in {0..7,16..23}; lanes 16-31 (M=l-16): K in {8..15,24..31}.
__device__ __forceinline__ v16bf load_a_frag(const __bf16* act, int kt, int mrow, int khalf) {
  const __bf16* p = act + mrow * H_ + kt * 32 + khalf * 8;
  v8bf lo = *(const v8bf*)p;
  v8bf hi = *(const v8bf*)(p + 16);
  v16bf a;
#pragma unroll
  for (int e = 0; e < 8; ++e) { a[e] = lo[e]; a[e + 8] = hi[e]; }
  return a;
}

__device__ __forceinline__ v16bf load_b_frag(const __bf16* pw, int kt, int nt, int lane, int ntiles) {
  size_t off = (((size_t)(kt * ntiles + nt)) * 32 + lane) * 16;
  return *(const v16bf*)(pw + off);
}

__device__ __forceinline__ v8f wmma_bf16(v16bf a, v16bf b, v8f c) {
  return __builtin_amdgcn_wmma_f32_16x16x32_bf16(
      /*neg_a=*/false, a, /*neg_b=*/false, b,
      /*c_mod=*/(short)0, c, /*reuse_a=*/false, /*reuse_b=*/false);
}

__device__ __forceinline__ float sigmoidf_(float x) {
  return 1.0f / (1.0f + __expf(-x));
}

// ---------------------------------------------------------------------------
// Main persistent GRU kernel
// ---------------------------------------------------------------------------
__global__ void __launch_bounds__(512)
gru_scan_kernel(const float* __restrict__ input,    // [B,S,I] f32
                const float* __restrict__ h0,       // [B,L,H] f32
                const float* __restrict__ bz,       // [L,H]
                const float* __restrict__ br,       // [L,H]
                const float* __restrict__ bg,       // [L,H]
                const float* __restrict__ by,       // [O]
                const __bf16* __restrict__ pw,      // packed weights
                __bf16* __restrict__ hstate,        // [L,B,H] bf16 master
                float* __restrict__ out)            // [B*S*O] ++ [B*L*H]
{
  __shared__ __bf16 xact [16 * H_];   // layer input (A operand)
  __shared__ __bf16 hbuf [16 * H_];   // current-layer h (async-staged)
  __shared__ __bf16 zbuf [16 * H_];   // z gate
  __shared__ __bf16 rhbuf[16 * H_];   // r * h (A operand for g)

  const int tid   = threadIdx.x;
  const int lane  = tid & 31;
  const int wv    = tid >> 5;         // 0..15
  const int m0    = blockIdx.x * 16;  // batch-row base
  const int mrow  = lane & 15;
  const int khalf = lane >> 4;

  const __bf16* pWhy = pw + WHY_OFF;
  // Generic LDS pointers truncate to the LDS byte offset in the low 32 bits
  // (ISA 10.2 aperture rules) -> usable as async-load VDST LDS address.
  const unsigned hbuf_lds = (unsigned)(uintptr_t)&hbuf[0];

  // Initialize bf16 master hidden state from h0 [B,L,H] -> hstate [L,B,H].
  for (int i = tid; i < L_ * 16 * H_; i += blockDim.x) {
    int k = i / (16 * H_);
    int r = i % (16 * H_);
    int m = r / H_, hh = r % H_;
    hstate[((size_t)k * B_ + m0 + m) * H_ + hh] =
        (__bf16)h0[((size_t)(m0 + m) * L_ + k) * H_ + hh];
  }
  __syncthreads();

#pragma unroll 1
  for (int t = 0; t < S_; ++t) {
    // Stage layer-0 input x_t (f32 -> bf16 LDS, row-major [16][512]).
    for (int i = tid; i < 16 * H_; i += blockDim.x) {
      int m = i >> 9, c = i & 511;
      xact[i] = (__bf16)input[((size_t)(m0 + m) * S_ + t) * I_ + c];
    }
    __syncthreads();

#pragma unroll 1
    for (int k = 0; k < L_; ++k) {
      // ---- async DMA stage of h_k into LDS (16 KB contiguous) ----
      {
        const __bf16* gsrc = hstate + ((size_t)k * B_ + m0) * H_;
#pragma unroll
        for (int i = 0; i < 2; ++i) {            // 1024 x 16B chunks / 512 thr
          int chunk = tid + i * 512;
          unsigned    ldst = hbuf_lds + chunk * 16;
          const void* gptr = (const void*)(gsrc + chunk * 8);
          asm volatile("global_load_async_to_lds_b128 %0, %1, off"
                       :: "v"(ldst), "v"(gptr) : "memory");
        }
        asm volatile("s_wait_asynccnt 0x0" ::: "memory");
      }
      __syncthreads();

      const __bf16* pWxz = pw + (size_t)(0 * L_ + k) * WELEMS_PER;
      const __bf16* pWhz = pw + (size_t)(1 * L_ + k) * WELEMS_PER;
      const __bf16* pWxr = pw + (size_t)(2 * L_ + k) * WELEMS_PER;
      const __bf16* pWhr = pw + (size_t)(3 * L_ + k) * WELEMS_PER;
      const __bf16* pWxg = pw + (size_t)(4 * L_ + k) * WELEMS_PER;
      const __bf16* pWhg = pw + (size_t)(5 * L_ + k) * WELEMS_PER;

      // ---- z & r pass: 2 nt columns per wave, z/r chains share A frags ----
#pragma unroll
      for (int i = 0; i < 2; ++i) {
        int nt = wv * 2 + i;                     // 16 waves x 2 = 32 columns
        v8f cz = {}, cr = {};
        for (int kt = 0; kt < 16; ++kt) {
          if (kt < 15)
            __builtin_prefetch(pWxz + (((size_t)((kt + 1) * 32 + nt)) * 32 + lane) * 16, 0, 3);
          v16bf ax  = load_a_frag(xact, kt, mrow, khalf);
          v16bf ah  = load_a_frag(hbuf, kt, mrow, khalf);
          v16bf bxz = load_b_frag(pWxz, kt, nt, lane, 32);
          v16bf bxr = load_b_frag(pWxr, kt, nt, lane, 32);
          v16bf bhz = load_b_frag(pWhz, kt, nt, lane, 32);
          v16bf bhr = load_b_frag(pWhr, kt, nt, lane, 32);
          cz = wmma_bf16(ax, bxz, cz);           // two independent chains:
          cr = wmma_bf16(ax, bxr, cr);           // hazard slots overlap
          cz = wmma_bf16(ah, bhz, cz);
          cr = wmma_bf16(ah, bhr, cr);
        }
        int   n   = nt * 16 + mrow;
        float bzv = bz[k * H_ + n];
        float brv = br[k * H_ + n];
#pragma unroll
        for (int q = 0; q < 8; ++q) {
          int   m  = khalf * 8 + q;              // C/D rows: lanes<16 -> q, else 8+q
          float zv = sigmoidf_(cz[q] + bzv);
          float rv = sigmoidf_(cr[q] + brv);
          float hv = (float)hbuf[m * H_ + n];
          zbuf [m * H_ + n] = (__bf16)zv;
          rhbuf[m * H_ + n] = (__bf16)(rv * hv);
        }
      }
      __syncthreads();  // zbuf / rhbuf ready

      // ---- g pass: both nt tiles share A frags (xact, rhbuf) ----
      const int nt0 = wv * 2, nt1 = wv * 2 + 1;
      v8f c0 = {}, c1 = {};
      for (int kt = 0; kt < 16; ++kt) {
        if (kt < 15)
          __builtin_prefetch(pWxg + (((size_t)((kt + 1) * 32 + nt0)) * 32 + lane) * 16, 0, 3);
        v16bf ax = load_a_frag(xact,  kt, mrow, khalf);
        v16bf ar = load_a_frag(rhbuf, kt, mrow, khalf);
        v16bf bx0 = load_b_frag(pWxg, kt, nt0, lane, 32);
        v16bf bx1 = load_b_frag(pWxg, kt, nt1, lane, 32);
        v16bf bh0 = load_b_frag(pWhg, kt, nt0, lane, 32);
        v16bf bh1 = load_b_frag(pWhg, kt, nt1, lane, 32);
        c0 = wmma_bf16(ax, bx0, c0);
        c1 = wmma_bf16(ax, bx1, c1);
        c0 = wmma_bf16(ar, bh0, c0);
        c1 = wmma_bf16(ar, bh1, c1);
      }

      v8f gacc[2] = {c0, c1};
      float hn[2][8];
#pragma unroll
      for (int i = 0; i < 2; ++i) {
        int   nt  = wv * 2 + i;
        int   n   = nt * 16 + mrow;
        float bgv = bg[k * H_ + n];
#pragma unroll
        for (int q = 0; q < 8; ++q) {
          int   m  = khalf * 8 + q;
          float gv = tanhf(gacc[i][q] + bgv);
          float zv = (float)zbuf[m * H_ + n];
          float hv = (float)hbuf[m * H_ + n];
          hn[i][q] = zv * hv + (1.0f - zv) * gv;
        }
      }
      __syncthreads();  // all reads of xact/hbuf/zbuf/rhbuf complete

      // h' -> next layer input (in place) + bf16 master copy.
#pragma unroll
      for (int i = 0; i < 2; ++i) {
        int nt = wv * 2 + i;
        int n  = nt * 16 + mrow;
#pragma unroll
        for (int q = 0; q < 8; ++q) {
          int m = khalf * 8 + q;
          xact[m * H_ + n] = (__bf16)hn[i][q];
          hstate[((size_t)k * B_ + m0 + m) * H_ + n] = (__bf16)hn[i][q];
        }
      }
      __syncthreads();
    }  // layers

    // ---- output projection: y[b,t,:] = h2 @ Why + by (waves 0..7) ----
    if (wv < 8) {  // wave-uniform: EXEC all-ones inside
      int nt = wv;
      v8f ce = {}, co = {};
      for (int kt = 0; kt < 16; kt += 2) {       // even/odd chains for ILP
        v16bf a0 = load_a_frag(xact, kt, mrow, khalf);
        v16bf b0 = load_b_frag(pWhy, kt, nt, lane, 8);
        v16bf a1 = load_a_frag(xact, kt + 1, mrow, khalf);
        v16bf b1 = load_b_frag(pWhy, kt + 1, nt, lane, 8);
        ce = wmma_bf16(a0, b0, ce);
        co = wmma_bf16(a1, b1, co);
      }
      int   n    = nt * 16 + mrow;
      float bias = by[n];
#pragma unroll
      for (int q = 0; q < 8; ++q) {
        int m = khalf * 8 + q;
        out[((size_t)(m0 + m) * S_ + t) * O_ + n] = ce[q] + co[q] + bias;
      }
    }
    __syncthreads();  // before next step overwrites xact
  }  // timesteps

  // ---- final hidden state: out2[b,l,h] = hstate[l,b,h] ----
  for (int i = tid; i < L_ * 16 * H_; i += blockDim.x) {
    int k = i / (16 * H_);
    int r = i % (16 * H_);
    int m = r / H_, hh = r % H_;
    out[(size_t)B_ * S_ * O_ + ((size_t)(m0 + m) * L_ + k) * H_ + hh] =
        (float)hstate[((size_t)k * B_ + m0 + m) * H_ + hh];
  }
}

// ---------------------------------------------------------------------------
extern "C" void kernel_launch(void* const* d_in, const int* in_sizes, int n_in,
                              void* d_out, int out_size, void* d_ws, size_t ws_size,
                              hipStream_t stream) {
  (void)in_sizes; (void)n_in; (void)out_size; (void)ws_size;

  const float* input = (const float*)d_in[0];
  const float* h0    = (const float*)d_in[1];
  const float* Wxz   = (const float*)d_in[2];
  const float* Whz   = (const float*)d_in[3];
  const float* bz    = (const float*)d_in[4];
  const float* Wxr   = (const float*)d_in[5];
  const float* Whr   = (const float*)d_in[6];
  const float* br    = (const float*)d_in[7];
  const float* Wxg   = (const float*)d_in[8];
  const float* Whg   = (const float*)d_in[9];
  const float* bg    = (const float*)d_in[10];
  const float* Why   = (const float*)d_in[11];
  const float* by    = (const float*)d_in[12];

  __bf16* pw     = (__bf16*)d_ws;
  __bf16* hstate = (__bf16*)((char*)d_ws + HSTATE_OFF_BYTES);
  float*  out    = (float*)d_out;

  const int pthreads = 256;
  const unsigned pblocks = (unsigned)((PACK_TOTAL + pthreads - 1) / pthreads);
  pack_weights_kernel<<<dim3(pblocks), dim3(pthreads), 0, stream>>>(
      Wxz, Whz, Wxr, Whr, Wxg, Whg, Why, pw);

  gru_scan_kernel<<<dim3(B_ / 16), dim3(512), 0, stream>>>(
      input, h0, bz, br, bg, by, pw, hstate, out);
}
